// MLPBase_47596827574897
// MI455X (gfx1250) — matrix operand order repaired and checked by reading, not compiled, not run on hardware
//
#include <hip/hip_runtime.h>
#include <hip/hip_bf16.h>
#include <stddef.h>
#include <stdint.h>

// ---------------------------------------------------------------- constants
#define BB    256          // batch
#define TT    512          // time steps
#define DD    512          // input dim
#define HH    512          // hidden dim
#define G3    1536         // 3*H
#define DICT  1024
#define QQ    32
#define OUTD  (DICT*QQ)    // 32768

typedef __bf16 bf16_t;
typedef __attribute__((ext_vector_type(16))) __bf16 v16bf;
typedef __attribute__((ext_vector_type(8)))  __bf16 v8bf;
typedef __attribute__((ext_vector_type(8)))  float  v8f;

// workspace layout (bytes)
#define OFF_XB    ((size_t)0)                            // B*T*D bf16 (134 MB)
#define OFF_WIH   (OFF_XB  + (size_t)BB*TT*DD*2)         // 1536*512 bf16
#define OFF_WHH   (OFF_WIH + (size_t)G3*DD*2)            // 1536*512 bf16
#define OFF_WOUT  (OFF_WHH + (size_t)G3*HH*2)            // 32768*512 bf16
#define OFF_H0    (OFF_WOUT + (size_t)OUTD*HH*2)         // 256*512 f32
#define OFF_H1    (OFF_H0  + (size_t)BB*HH*4)
#define OFF_HB0   (OFF_H1  + (size_t)BB*HH*4)            // 256*512 bf16
#define OFF_HB1   (OFF_HB0 + (size_t)BB*HH*2)
#define OFF_LOG   (OFF_HB1 + (size_t)BB*HH*2)            // 256*32768 f32
#define OFF_C1    (OFF_LOG + (size_t)BB*OUTD*4)          // 256*512 f32
#define OFF_C2    (OFF_C1  + (size_t)BB*HH*4)            // 256*512 f32

// ---------------------------------------------------------------- helpers
__device__ inline bf16_t f2bf(float f) {
  union { float f; unsigned u; } c; c.f = f;
  unsigned r = (c.u + 0x7FFFu + ((c.u >> 16) & 1u)) >> 16;
  unsigned short s = (unsigned short)r;
  return __builtin_bit_cast(bf16_t, s);
}

__device__ inline float sigmoidf_(float x) { return 1.0f / (1.0f + __expf(-x)); }

__device__ inline unsigned hash_u32(unsigned x) {
  x ^= x >> 16; x *= 0x7feb352dU;
  x ^= x >> 15; x *= 0x846ca68bU;
  x ^= x >> 16; return x;
}

// gfx1250 async copy: 16 bytes global -> LDS, tracked by ASYNCcnt.
// GVS mode: mem_addr = SADDR + VADDR(voff) ; VDST = LDS byte address.
__device__ inline void async_copy_b128(unsigned lds_addr, const void* sbase, int voff) {
  asm volatile("global_load_async_to_lds_b128 %0, %1, %2"
               :
               : "v"(lds_addr), "v"(voff), "s"(sbase)
               : "memory");
}
__device__ inline void async_wait0() {
  asm volatile("s_wait_asynccnt 0" ::: "memory");
}
__device__ inline unsigned lds_addr_of(const void* p) {
  // Generic pointers to LDS carry the workgroup-relative LDS offset in the
  // low 32 bits.
  return (unsigned)(uintptr_t)p;
}

// A-fragment (16x32 bf16, MxK) from a 16x512 row-major LDS tile.
// Layout: lane L<16 -> M=L, K = {0..15}; lane L>=16 -> M=L-16,
// K = {16..31}. One contiguous 32B run per lane (two 16B loads).
__device__ inline v16bf load_a_frag(const bf16_t* s, int lane, int k0) {
  const int m = lane & 15, half = lane >> 4;
  const int base = m * 512 + k0 + half * 16;
  v8bf lo = *(const v8bf*)(s + base);
  v8bf hi = *(const v8bf*)(s + base + 8);
  return __builtin_shufflevector(lo, hi, 0,1,2,3,4,5,6,7,8,9,10,11,12,13,14,15);
}

// ---------------------------------------------------------------- kernels
__global__ void cvt_bf16_k(const float* __restrict__ src, bf16_t* __restrict__ dst, int n) {
  int i = blockIdx.x * 256 + threadIdx.x;
  if (i < n) dst[i] = f2bf(src[i]);
}

__global__ void zero_f_k(float* __restrict__ p, int n) {
  int i = blockIdx.x * 256 + threadIdx.x;
  if (i < n) p[i] = 0.0f;
}

// One GRU timestep: h_next = GRUCell(x_t, h_cur) with ragged-length freeze.
// grid = (H/128, B/16), block = 256 (8 waves). Wave w owns 16 cols of H.
// Staging of the x_t and h bf16 tiles uses gfx1250 async global->LDS copies.
__global__ void gru_step_k(const bf16_t* __restrict__ xb,     // [B,T,D] bf16
                           const int*   __restrict__ lengths, // [B]
                           const bf16_t* __restrict__ Wih,    // [1536,512] bf16 row-major
                           const bf16_t* __restrict__ Whh,    // [1536,512] bf16 row-major
                           const float* __restrict__ b_ih,    // [1536]
                           const float* __restrict__ b_hh,    // [1536]
                           const float* __restrict__ h_cur,   // [B,H] f32
                           const bf16_t* __restrict__ hb_cur, // [B,H] bf16
                           float*       __restrict__ h_next,  // [B,H] f32
                           bf16_t*      __restrict__ hb_next, // [B,H] bf16
                           int t) {
  __shared__ __align__(16) bf16_t sX[16 * 512];
  __shared__ __align__(16) bf16_t sH[16 * 512];

  const int m0   = blockIdx.y * 16;
  const int nblk = blockIdx.x * 128;
  const int tid  = threadIdx.x;
  const int wave = tid >> 5;
  const int lane = tid & 31;

  // Async-stage 16 rows (1 KB each) of x_t and h into LDS.
  // 64 x 16B transfers per row; 1024 transfers per tile.
  {
    const bf16_t* xbase = xb + ((size_t)m0 * TT + t) * DD;  // row m -> +m*T*D
    const bf16_t* hbase = hb_cur + (size_t)m0 * HH;
    for (int j = tid; j < 1024; j += 256) {
      const int m   = j >> 6;
      const int off = (j & 63) * 16;
      async_copy_b128(lds_addr_of((const char*)sX + j * 16),
                      xbase, m * (TT * DD * 2) + off);
      async_copy_b128(lds_addr_of((const char*)sH + j * 16),
                      hbase, m * (HH * 2) + off);
    }
  }
  async_wait0();
  __syncthreads();

  const int half = lane >> 4;
  const int nl   = lane & 15;
  const int ncol = nblk + wave * 16 + nl;   // column of H this lane produces

  v8f acc[6] = {};   // i_r, i_z, i_n, h_r, h_z, h_n

  for (int k0 = 0; k0 < 512; k0 += 32) {
    v16bf ax = load_a_frag(sX, lane, k0);
    v16bf ah = load_a_frag(sH, lane, k0);
#pragma unroll
    for (int g = 0; g < 3; ++g) {
      const size_t grow = (size_t)(g * 512 + ncol);
      v16bf bi = *(const v16bf*)(Wih + grow * 512 + k0 + half * 16);
      acc[g] = __builtin_amdgcn_wmma_f32_16x16x32_bf16(
                   false, ax, false, bi, (short)0, acc[g], false, false);
      v16bf bh = *(const v16bf*)(Whh + grow * 512 + k0 + half * 16);
      acc[3 + g] = __builtin_amdgcn_wmma_f32_16x16x32_bf16(
                   false, ah, false, bh, (short)0, acc[3 + g], false, false);
    }
  }

  const float bir = b_ih[ncol],        bhr = b_hh[ncol];
  const float biz = b_ih[512 + ncol],  bhz = b_hh[512 + ncol];
  const float bin = b_ih[1024 + ncol], bhn = b_hh[1024 + ncol];

#pragma unroll
  for (int e = 0; e < 8; ++e) {
    const int m = m0 + e + 8 * half;
    const float r  = sigmoidf_(acc[0][e] + bir + acc[3][e] + bhr);
    const float z  = sigmoidf_(acc[1][e] + biz + acc[4][e] + bhz);
    const float nn = tanhf(acc[2][e] + bin + r * (acc[5][e] + bhn));
    const float hold = h_cur[(size_t)m * HH + ncol];
    const float hnew = (1.0f - z) * nn + z * hold;
    const float outv = (t < lengths[m]) ? hnew : hold;
    h_next [(size_t)m * HH + ncol] = outv;
    hb_next[(size_t)m * HH + ncol] = f2bf(outv);
  }
}

// Actor head: logits[m, col] = hF[m,:] . W_out[col,:] + b_out[col]
// grid = (OUTD/128, B/16), block = 256 (8 waves, one 16x16 tile each).
__global__ void actor_gemm_k(const bf16_t* __restrict__ hbF,   // [B,H] bf16
                             const bf16_t* __restrict__ Wb,    // [32768,512] bf16
                             const float* __restrict__ bias,   // [32768]
                             float* __restrict__ outL) {       // [B,32768]
  __shared__ __align__(16) bf16_t sH[16 * 512];
  const int m0  = blockIdx.y * 16;
  const int n0  = blockIdx.x * 128;
  const int tid = threadIdx.x;
  const int wave = tid >> 5, lane = tid & 31;

  {
    const bf16_t* hbase = hbF + (size_t)m0 * HH;
    for (int j = tid; j < 1024; j += 256) {
      const int m   = j >> 6;
      const int off = (j & 63) * 16;
      async_copy_b128(lds_addr_of((const char*)sH + j * 16),
                      hbase, m * (HH * 2) + off);
    }
  }
  async_wait0();
  __syncthreads();

  const int half = lane >> 4, nl = lane & 15;
  const int col  = n0 + wave * 16 + nl;

  v8f acc = {};
  for (int k0 = 0; k0 < 512; k0 += 32) {
    v16bf a = load_a_frag(sH, lane, k0);
    v16bf b = *(const v16bf*)(Wb + (size_t)col * 512 + k0 + half * 16);
    acc = __builtin_amdgcn_wmma_f32_16x16x32_bf16(
              false, a, false, b, (short)0, acc, false, false);
  }
  const float bb = bias[col];
#pragma unroll
  for (int e = 0; e < 8; ++e) {
    const int m = m0 + e + 8 * half;
    outL[(size_t)m * OUTD + col] = acc[e] + bb;
  }
}

// Per (b,q) row of 1024 logits: log-softmax + Gumbel-argmax sample + one-hot.
__global__ void softmax_sample_k(const float* __restrict__ logits, // [B*Q, DICT]
                                 float* __restrict__ q_out,        // [B*Q, DICT]
                                 float* __restrict__ lp_out) {     // [B*Q, DICT]
  const int row = blockIdx.x;
  const int tid = threadIdx.x;
  const float* lp = logits + (size_t)row * DICT;

  float v[4];
#pragma unroll
  for (int i = 0; i < 4; ++i) v[i] = lp[tid + 256 * i];

  __shared__ float red[256];
  __shared__ int   redi[256];

  float m = fmaxf(fmaxf(v[0], v[1]), fmaxf(v[2], v[3]));
  red[tid] = m; __syncthreads();
  for (int s = 128; s > 0; s >>= 1) {
    if (tid < s) red[tid] = fmaxf(red[tid], red[tid + s]);
    __syncthreads();
  }
  const float mx = red[0]; __syncthreads();

  float se = 0.0f;
#pragma unroll
  for (int i = 0; i < 4; ++i) se += __expf(v[i] - mx);
  red[tid] = se; __syncthreads();
  for (int s = 128; s > 0; s >>= 1) {
    if (tid < s) red[tid] += red[tid + s];
    __syncthreads();
  }
  const float lz = mx + __logf(red[0]); __syncthreads();

  // Gumbel-argmax categorical draw (hash RNG stands in for threefry).
  float best = -1e30f; int bidx = 0;
#pragma unroll
  for (int i = 0; i < 4; ++i) {
    const int d = tid + 256 * i;
    const unsigned h = hash_u32((unsigned)(row * DICT + d) ^ 0x9E3779B9u);
    const float u = (float)(h >> 8) * (1.0f / 16777216.0f) + 1e-12f;
    const float g = -__logf(-__logf(u));
    const float sc = (v[i] - lz) + g;
    if (sc > best) { best = sc; bidx = d; }
  }
  red[tid] = best; redi[tid] = bidx; __syncthreads();
  for (int s = 128; s > 0; s >>= 1) {
    if (tid < s && red[tid + s] > red[tid]) {
      red[tid] = red[tid + s]; redi[tid] = redi[tid + s];
    }
    __syncthreads();
  }
  const int sel = redi[0];

#pragma unroll
  for (int i = 0; i < 4; ++i) {
    const int d = tid + 256 * i;
    lp_out[(size_t)row * DICT + d] = v[i] - lz;
    q_out [(size_t)row * DICT + d] = (d == sel) ? 1.0f : 0.0f;
  }
}

// Critic linear (+optional ReLU): out[b,j] = act(in[b,:] . W[j,:] + bias[j])
__global__ void linrelu_k(const float* __restrict__ in, const float* __restrict__ W,
                          const float* __restrict__ bias, float* __restrict__ out,
                          int do_relu) {
  const int j  = blockIdx.x * 256 + threadIdx.x;  // 0..H-1
  const int b  = blockIdx.y;
  const float* ip = in + (size_t)b * HH;
  const float* wp = W + (size_t)j * HH;
  float s = bias[j];
  for (int k = 0; k < HH; ++k) s += ip[k] * wp[k];
  if (do_relu) s = fmaxf(s, 0.0f);
  out[(size_t)b * HH + j] = s;
}

__global__ void value_k(const float* __restrict__ h2, const float* __restrict__ Wc3,
                        const float* __restrict__ bc3, float* __restrict__ out) {
  const int b = blockIdx.x;
  __shared__ float red[256];
  float s = 0.0f;
  for (int k = threadIdx.x; k < HH; k += 256) s += h2[(size_t)b * HH + k] * Wc3[k];
  red[threadIdx.x] = s; __syncthreads();
  for (int st = 128; st > 0; st >>= 1) {
    if (threadIdx.x < st) red[threadIdx.x] += red[threadIdx.x + st];
    __syncthreads();
  }
  if (threadIdx.x == 0) out[b] = red[0] + bc3[0];
}

// ---------------------------------------------------------------- launch
extern "C" void kernel_launch(void* const* d_in, const int* in_sizes, int n_in,
                              void* d_out, int out_size, void* d_ws, size_t ws_size,
                              hipStream_t stream) {
  (void)in_sizes; (void)n_in; (void)out_size; (void)ws_size;

  const float* x       = (const float*)d_in[0];
  const int*   lengths = (const int*)  d_in[1];
  const float* W_ih    = (const float*)d_in[2];
  const float* W_hh    = (const float*)d_in[3];
  const float* b_ih    = (const float*)d_in[4];
  const float* b_hh    = (const float*)d_in[5];
  const float* W_out   = (const float*)d_in[6];
  const float* b_out   = (const float*)d_in[7];
  const float* Wc1     = (const float*)d_in[8];
  const float* bc1     = (const float*)d_in[9];
  const float* Wc2     = (const float*)d_in[10];
  const float* bc2     = (const float*)d_in[11];
  const float* Wc3     = (const float*)d_in[12];
  const float* bc3     = (const float*)d_in[13];

  float* out = (float*)d_out;
  char*  ws  = (char*)d_ws;

  bf16_t* xb     = (bf16_t*)(ws + OFF_XB);
  bf16_t* wih_b  = (bf16_t*)(ws + OFF_WIH);
  bf16_t* whh_b  = (bf16_t*)(ws + OFF_WHH);
  bf16_t* wout_b = (bf16_t*)(ws + OFF_WOUT);
  float*  h0     = (float*)(ws + OFF_H0);
  float*  h1     = (float*)(ws + OFF_H1);
  bf16_t* hb0    = (bf16_t*)(ws + OFF_HB0);
  bf16_t* hb1    = (bf16_t*)(ws + OFF_HB1);
  float*  logits = (float*)(ws + OFF_LOG);
  float*  c1     = (float*)(ws + OFF_C1);
  float*  c2     = (float*)(ws + OFF_C2);

  // 1) One-time bf16 conversions: x (read once, ~17us of HBM) + weights.
  {
    const int nx = BB * TT * DD;
    cvt_bf16_k<<<(nx + 255) / 256, 256, 0, stream>>>(x, xb, nx);
  }
  cvt_bf16_k<<<(G3 * DD + 255) / 256, 256, 0, stream>>>(W_ih, wih_b, G3 * DD);
  cvt_bf16_k<<<(G3 * HH + 255) / 256, 256, 0, stream>>>(W_hh, whh_b, G3 * HH);
  cvt_bf16_k<<<(OUTD * HH + 255) / 256, 256, 0, stream>>>(W_out, wout_b, OUTD * HH);

  // 2) h = 0 (f32 and bf16 views), then 512 sequential GRU steps (ping-pong).
  zero_f_k<<<(BB * HH + 255) / 256, 256, 0, stream>>>(h0, BB * HH);
  zero_f_k<<<(BB * HH / 2 + 255) / 256, 256, 0, stream>>>((float*)hb0, BB * HH / 2);
  for (int t = 0; t < TT; ++t) {
    float*  hc  = (t & 1) ? h1  : h0;
    float*  hn  = (t & 1) ? h0  : h1;
    bf16_t* hbc = (t & 1) ? hb1 : hb0;
    bf16_t* hbn = (t & 1) ? hb0 : hb1;
    gru_step_k<<<dim3(HH / 128, BB / 16), 256, 0, stream>>>(
        xb, lengths, wih_b, whh_b, b_ih, b_hh, hc, hbc, hn, hbn, t);
  }
  float*  hF  = h0;    // T even -> final hidden lands in buffer 0
  bf16_t* hbF = hb0;

  // 3) Actor logits GEMM + fused log-softmax / sample / one-hot.
  actor_gemm_k<<<dim3(OUTD / 128, BB / 16), 256, 0, stream>>>(hbF, wout_b, b_out, logits);
  float* q_out  = out + BB;                        // [B,Q,DICT]
  float* lp_out = out + BB + (size_t)BB * OUTD;    // [B,Q,DICT]
  softmax_sample_k<<<BB * QQ, 256, 0, stream>>>(logits, q_out, lp_out);

  // 4) Critic MLP -> value at out[0..B).
  linrelu_k<<<dim3(HH / 256, BB), 256, 0, stream>>>(hF, Wc1, bc1, c1, 1);
  linrelu_k<<<dim3(HH / 256, BB), 256, 0, stream>>>(c1, Wc2, bc2, c2, 1);
  value_k<<<BB, 256, 0, stream>>>(c2, Wc3, bc3, out);
}
